// ImageEncoderViT_3272765080383
// MI455X (gfx1250) — matrix operand order repaired and checked by reading, not compiled, not run on hardware
//
#include <hip/hip_runtime.h>
#include <math.h>

// ---------------------------------------------------------------------------
// SAM ViT image encoder for gfx1250 (MI455X). All GEMMs use
// v_wmma_f32_16x16x32_bf16. 64x64 workgroup tile, 8 wave32s each doing a
// 16x32 slab (2 WMMAs / K-step). Global->register->LDS software pipeline
// with double-buffered LDS so global latency hides behind the WMMAs.
// ---------------------------------------------------------------------------

typedef __attribute__((ext_vector_type(16))) __bf16 bf16x16;
typedef __attribute__((ext_vector_type(4)))  __bf16 bf16x4;
typedef __attribute__((ext_vector_type(8)))  float  f32x8;

#define DD    768
#define HEADS 12
#define HDIM  64
#define MLPD  3072
#define OUTC  256
#define GRIDS 32
#define WIN   14
#define NTOK  2048          // 2 * 32 * 32
#define TWIN  3528          // 18 windows * 196
#define SCALE_ATT 0.125f    // 64^-0.5

#define BM 64
#define BN 64
#define TK 32
#define LDSS 40             // LDS row stride (bf16 elems), padded vs 32

// ------------------------- generic batched WMMA GEMM ------------------------
// C[z] = act( alpha * A[z] @ op(B[z]) + bias (+ resid) )
// batch offset: z -> (z/heads)*sw + (z%heads)*sh
__global__ __launch_bounds__(256)
void gemm_bf16_wmma(const float* __restrict__ A, int lda, long long sAw, long long sAh,
                    const float* __restrict__ B, int ldb, long long sBw, long long sBh, int bT,
                    float* __restrict__ C, int ldc, long long sCw, long long sCh,
                    const float* __restrict__ bias,
                    const float* __restrict__ resid, int ldr,
                    int M, int N, int K, int heads, float alpha, int act)
{
  __shared__ __bf16 As[2][BM][LDSS];
  __shared__ __bf16 Bs[2][BN][LDSS];

  const int z  = blockIdx.z;
  const int zw = z / heads, zh = z % heads;
  A += (long long)zw * sAw + (long long)zh * sAh;
  B += (long long)zw * sBw + (long long)zh * sBh;
  C += (long long)zw * sCw + (long long)zh * sCh;

  const int m0   = blockIdx.y * BM;
  const int n0   = blockIdx.x * BN;
  const int tid  = threadIdx.x;
  const int lane = tid & 31, wid = tid >> 5;
  const int wm   = wid >> 1, wn = wid & 1;     // 4 x 2 waves -> 64x64 tile
  const int hi   = lane >> 4, lm = lane & 15;

  // staging coordinates (A: rows a_mi and a_mi+32; B: two rows / k-groups)
  const int a_mi = tid >> 3;             // 0..31
  const int a_kk = (tid & 7) << 2;       // 0,4,...,28
  const int bt_nn = tid >> 3;            // bT:  rows nn, nn+32
  const int bt_kk = (tid & 7) << 2;
  const int bn_kk = tid >> 4;            // !bT: k rows kk, kk+16
  const int bn_nn = (tid & 15) << 2;

  f32x8 acc0 = {0.f, 0.f, 0.f, 0.f, 0.f, 0.f, 0.f, 0.f};
  f32x8 acc1 = {0.f, 0.f, 0.f, 0.f, 0.f, 0.f, 0.f, 0.f};

  float4 ra0, ra1, rb0, rb1;

  auto load_tile = [&](int k0, float4& a0, float4& a1, float4& b0, float4& b1) {
    { // ---- A: 64 x 32
      const bool fast = ((m0 + BM) <= M) && ((k0 + TK) <= K);
      if (fast) {
        a0 = *(const float4*)(A + (long long)(m0 + a_mi) * lda + k0 + a_kk);
        a1 = *(const float4*)(A + (long long)(m0 + a_mi + 32) * lda + k0 + a_kk);
      } else {
        const bool ok0 = (m0 + a_mi) < M, ok1 = (m0 + a_mi + 32) < M;
        const int gm0 = min(m0 + a_mi, M - 1), gm1 = min(m0 + a_mi + 32, M - 1);
        #pragma unroll
        for (int j = 0; j < 4; ++j) {
          int gk = k0 + a_kk + j;
          int gkc = min(gk, K - 1);
          float v0 = A[(long long)gm0 * lda + gkc];
          float v1 = A[(long long)gm1 * lda + gkc];
          if (!ok0 || gk >= K) v0 = 0.f;
          if (!ok1 || gk >= K) v1 = 0.f;
          (&a0.x)[j] = v0;
          (&a1.x)[j] = v1;
        }
      }
    }
    if (bT) { // ---- B is N x K row-major: rows n, n+32 ; 4 consecutive k
      const bool fast = ((n0 + BN) <= N) && ((k0 + TK) <= K);
      if (fast) {
        b0 = *(const float4*)(B + (long long)(n0 + bt_nn) * ldb + k0 + bt_kk);
        b1 = *(const float4*)(B + (long long)(n0 + bt_nn + 32) * ldb + k0 + bt_kk);
      } else {
        const bool ok0 = (n0 + bt_nn) < N, ok1 = (n0 + bt_nn + 32) < N;
        const int gn0 = min(n0 + bt_nn, N - 1), gn1 = min(n0 + bt_nn + 32, N - 1);
        #pragma unroll
        for (int j = 0; j < 4; ++j) {
          int gk = k0 + bt_kk + j;
          int gkc = min(gk, K - 1);
          float v0 = B[(long long)gn0 * ldb + gkc];
          float v1 = B[(long long)gn1 * ldb + gkc];
          if (!ok0 || gk >= K) v0 = 0.f;
          if (!ok1 || gk >= K) v1 = 0.f;
          (&b0.x)[j] = v0;
          (&b1.x)[j] = v1;
        }
      }
    } else { // ---- B is K x N row-major: k rows kk, kk+16 ; 4 consecutive n
      const bool fast = ((n0 + BN) <= N) && ((k0 + TK) <= K);
      if (fast) {
        b0 = *(const float4*)(B + (long long)(k0 + bn_kk) * ldb + n0 + bn_nn);
        b1 = *(const float4*)(B + (long long)(k0 + bn_kk + 16) * ldb + n0 + bn_nn);
      } else {
        const bool ok0 = (k0 + bn_kk) < K, ok1 = (k0 + bn_kk + 16) < K;
        const int gk0 = min(k0 + bn_kk, K - 1), gk1 = min(k0 + bn_kk + 16, K - 1);
        #pragma unroll
        for (int j = 0; j < 4; ++j) {
          int gn = n0 + bn_nn + j;
          int gnc = min(gn, N - 1);
          float v0 = B[(long long)gk0 * ldb + gnc];
          float v1 = B[(long long)gk1 * ldb + gnc];
          if (!ok0 || gn >= N) v0 = 0.f;
          if (!ok1 || gn >= N) v1 = 0.f;
          (&b0.x)[j] = v0;
          (&b1.x)[j] = v1;
        }
      }
    }
  };

  auto store_tile = [&](int buf) {
    bf16x4 o;
    o[0] = (__bf16)ra0.x; o[1] = (__bf16)ra0.y; o[2] = (__bf16)ra0.z; o[3] = (__bf16)ra0.w;
    *(bf16x4*)&As[buf][a_mi][a_kk] = o;
    o[0] = (__bf16)ra1.x; o[1] = (__bf16)ra1.y; o[2] = (__bf16)ra1.z; o[3] = (__bf16)ra1.w;
    *(bf16x4*)&As[buf][a_mi + 32][a_kk] = o;
    if (bT) {
      o[0] = (__bf16)rb0.x; o[1] = (__bf16)rb0.y; o[2] = (__bf16)rb0.z; o[3] = (__bf16)rb0.w;
      *(bf16x4*)&Bs[buf][bt_nn][bt_kk] = o;
      o[0] = (__bf16)rb1.x; o[1] = (__bf16)rb1.y; o[2] = (__bf16)rb1.z; o[3] = (__bf16)rb1.w;
      *(bf16x4*)&Bs[buf][bt_nn + 32][bt_kk] = o;
    } else { // transpose into Bs[n][k]
      Bs[buf][bn_nn + 0][bn_kk] = (__bf16)rb0.x;
      Bs[buf][bn_nn + 1][bn_kk] = (__bf16)rb0.y;
      Bs[buf][bn_nn + 2][bn_kk] = (__bf16)rb0.z;
      Bs[buf][bn_nn + 3][bn_kk] = (__bf16)rb0.w;
      Bs[buf][bn_nn + 0][bn_kk + 16] = (__bf16)rb1.x;
      Bs[buf][bn_nn + 1][bn_kk + 16] = (__bf16)rb1.y;
      Bs[buf][bn_nn + 2][bn_kk + 16] = (__bf16)rb1.z;
      Bs[buf][bn_nn + 3][bn_kk + 16] = (__bf16)rb1.w;
    }
  };

  const int nk = (K + TK - 1) / TK;
  load_tile(0, ra0, ra1, rb0, rb1);

  for (int t = 0; t < nk; ++t) {
    const int cur = t & 1;
    store_tile(cur);          // registers -> bf16 LDS (waits on loads of this tile)
    __syncthreads();

    if (t + 1 < nk)           // issue next tile's global loads; latency hides
      load_tile((t + 1) * TK, ra0, ra1, rb0, rb1);

    // A fragment: lanes 0-15 row m hold K 0-15 ; lanes 16-31 hold K 16-31
    bf16x16 af, bf0, bf1;
    const __bf16* ap = &As[cur][wm * 16 + lm][hi * 16];
    #pragma unroll
    for (int i = 0; i < 8; ++i) af[i] = ap[i];
    #pragma unroll
    for (int i = 0; i < 8; ++i) af[8 + i] = ap[8 + i];
    // B fragments: lanes 0-15 col n hold K 0-15 ; lanes 16-31 hold K 16-31
    const __bf16* bp0 = &Bs[cur][wn * 32 + lm][hi * 16];
    const __bf16* bp1 = &Bs[cur][wn * 32 + 16 + lm][hi * 16];
    #pragma unroll
    for (int i = 0; i < 16; ++i) bf0[i] = bp0[i];
    #pragma unroll
    for (int i = 0; i < 16; ++i) bf1[i] = bp1[i];

    acc0 = __builtin_amdgcn_wmma_f32_16x16x32_bf16(false, af, false, bf0,
                                                   (short)0, acc0, false, false);
    acc1 = __builtin_amdgcn_wmma_f32_16x16x32_bf16(false, af, false, bf1,
                                                   (short)0, acc1, false, false);
  }

  // C/D layout: VGPR r -> row = r + 8*hi, col = lane&15
  #pragma unroll
  for (int cg = 0; cg < 2; ++cg) {
    const int col = n0 + wn * 32 + cg * 16 + lm;
    if (col < N) {
      const float bv = bias ? bias[col] : 0.f;
      #pragma unroll
      for (int r = 0; r < 8; ++r) {
        int row = m0 + wm * 16 + hi * 8 + r;
        if (row < M) {
          float v = alpha * (cg ? acc1[r] : acc0[r]) + bv;
          if (resid) v += resid[(long long)row * ldr + col];
          if (act == 1) v = 0.5f * v * (1.f + erff(v * 0.70710678f)); // exact GELU
          C[(long long)row * ldc + col] = v;
        }
      }
    }
  }
}

// ------------------------- elementwise / reduction kernels ------------------

// im2col for 16x16/stride-16 patch conv: out (2048, 768)
__global__ __launch_bounds__(256)
void im2col_patch_kernel(const float* __restrict__ x, float* __restrict__ out)
{
  long long idx = (long long)blockIdx.x * blockDim.x + threadIdx.x;
  if (idx >= (long long)NTOK * DD) return;
  int t = (int)(idx / DD), k = (int)(idx % DD);
  int b = t >> 10, sp = t & 1023;
  int gy = sp >> 5, gx = sp & 31;
  int c = k >> 8, r = k & 255;
  int ph = r >> 4, pw = r & 15;
  out[idx] = x[(((long long)(b * 3 + c) * 512) + gy * 16 + ph) * 512 + gx * 16 + pw];
}

__global__ __launch_bounds__(256)
void add_pos_kernel(float* __restrict__ x, const float* __restrict__ pos)
{
  long long idx = (long long)blockIdx.x * blockDim.x + threadIdx.x;
  if (idx >= (long long)NTOK * DD) return;
  x[idx] += pos[idx % ((long long)1024 * DD)];
}

// row LayerNorm; out_nchw=1 writes final (B,C,32,32)
__global__ __launch_bounds__(256)
void ln_kernel(const float* __restrict__ in, float* __restrict__ out,
               const float* __restrict__ w, const float* __restrict__ b,
               int N, float eps, int out_nchw)
{
  __shared__ float red[256];
  const int row = blockIdx.x, tid = threadIdx.x;
  in += (long long)row * N;
  float s = 0.f;
  for (int i = tid; i < N; i += 256) s += in[i];
  red[tid] = s; __syncthreads();
  for (int o = 128; o > 0; o >>= 1) { if (tid < o) red[tid] += red[tid + o]; __syncthreads(); }
  const float mean = red[0] / N; __syncthreads();
  float v = 0.f;
  for (int i = tid; i < N; i += 256) { float d = in[i] - mean; v += d * d; }
  red[tid] = v; __syncthreads();
  for (int o = 128; o > 0; o >>= 1) { if (tid < o) red[tid] += red[tid + o]; __syncthreads(); }
  const float inv = rsqrtf(red[0] / N + eps);
  for (int i = tid; i < N; i += 256) {
    float val = (in[i] - mean) * inv * w[i] + b[i];
    if (!out_nchw) out[(long long)row * N + i] = val;
    else {
      int bb = row >> 10, sp = row & 1023;
      out[((long long)(bb * N + i) << 10) + sp] = val;
    }
  }
}

// pad + window partition: (2048,768) -> (18*196, 768), zeros in padding
__global__ __launch_bounds__(256)
void window_partition_kernel(const float* __restrict__ yln, float* __restrict__ out)
{
  long long idx = (long long)blockIdx.x * blockDim.x + threadIdx.x;
  if (idx >= (long long)TWIN * DD) return;
  int t = (int)(idx / DD), d = (int)(idx % DD);
  int win = t / 196, l = t % 196;
  int b = win / 9, wy = (win % 9) / 3, wx = win % 3;
  int ly = l / WIN, lx = l % WIN;
  int gy = wy * WIN + ly, gx = wx * WIN + lx;
  float v = 0.f;
  if (gy < GRIDS && gx < GRIDS)
    v = yln[((long long)(b << 10) + gy * GRIDS + gx) * DD + d];
  out[idx] = v;
}

// x += window_unpartition(proj) (or plain add when windowed==0)
__global__ __launch_bounds__(256)
void unpartition_add_kernel(float* __restrict__ x, const float* __restrict__ proj, int windowed)
{
  long long idx = (long long)blockIdx.x * blockDim.x + threadIdx.x;
  if (idx >= (long long)NTOK * DD) return;
  long long src = idx;
  if (windowed) {
    int t = (int)(idx / DD), d = (int)(idx % DD);
    int b = t >> 10, sp = t & 1023;
    int gy = sp >> 5, gx = sp & 31;
    int wy = gy / WIN, ly = gy % WIN, wx = gx / WIN, lx = gx % WIN;
    int win = b * 9 + wy * 3 + wx;
    src = ((long long)win * 196 + ly * WIN + lx) * DD + d;
  }
  x[idx] += proj[src];
}

// decomposed rel-pos: out[z,qi,j] = dot64(q[z,qi,:], table[coord(qi)-j+S-1,:])
__global__ __launch_bounds__(256)
void rel_kernel(const float* __restrict__ q, int ldq, long long sqw, long long sqh, int heads,
                const float* __restrict__ table, float* __restrict__ out,
                int HW, int Wq, int S, int isH, long long total)
{
  long long idx = (long long)blockIdx.x * blockDim.x + threadIdx.x;
  if (idx >= total) return;
  int j = (int)(idx % S);
  long long tmp = idx / S;
  int qi = (int)(tmp % HW);
  int z = (int)(tmp / HW);
  const float* qr = q + (long long)(z / heads) * sqw + (long long)(z % heads) * sqh
                      + (long long)qi * ldq;
  int coord = isH ? (qi / Wq) : (qi % Wq);
  const float* tr = table + (long long)(coord - j + S - 1) * HDIM;
  float s = 0.f;
  #pragma unroll
  for (int c = 0; c < HDIM; ++c) s += qr[c] * tr[c];
  out[idx] = s;
}

// scores[row] += rel_h[kh] + rel_w[kw]; softmax in place
__global__ __launch_bounds__(256)
void softmax_rel_kernel(float* __restrict__ scores, const float* __restrict__ relh,
                        const float* __restrict__ relw, int L, int Wk, int Hk)
{
  __shared__ float red[256];
  const long long rowid = blockIdx.x; // z*L + qi
  float* srow = scores + rowid * L;
  const float* rh = relh + rowid * Hk;
  const float* rw = relw + rowid * Wk;
  const int tid = threadIdx.x;
  float mx = -1e30f;
  for (int j = tid; j < L; j += 256) {
    float v = srow[j] + rh[j / Wk] + rw[j % Wk];
    srow[j] = v;
    mx = fmaxf(mx, v);
  }
  red[tid] = mx; __syncthreads();
  for (int o = 128; o > 0; o >>= 1) { if (tid < o) red[tid] = fmaxf(red[tid], red[tid + o]); __syncthreads(); }
  mx = red[0]; __syncthreads();
  float s = 0.f;
  for (int j = tid; j < L; j += 256) { float e = __expf(srow[j] - mx); srow[j] = e; s += e; }
  red[tid] = s; __syncthreads();
  for (int o = 128; o > 0; o >>= 1) { if (tid < o) red[tid] += red[tid + o]; __syncthreads(); }
  const float inv = 1.f / red[0];
  for (int j = tid; j < L; j += 256) srow[j] *= inv;
}

// im2col for neck 3x3 / pad 1 conv, from (2048, 256) token-major -> (2048, 2304)
__global__ __launch_bounds__(256)
void im2col_neck_kernel(const float* __restrict__ z, float* __restrict__ out)
{
  long long idx = (long long)blockIdx.x * blockDim.x + threadIdx.x;
  if (idx >= (long long)NTOK * (OUTC * 9)) return;
  int t = (int)(idx / (OUTC * 9)), k = (int)(idx % (OUTC * 9));
  int c = k / 9, r = k % 9;
  int kh = r / 3, kw = r % 3;
  int b = t >> 10, sp = t & 1023;
  int h = sp >> 5, wc = sp & 31;
  int hy = h + kh - 1, wx = wc + kw - 1;
  float v = 0.f;
  if (hy >= 0 && hy < GRIDS && wx >= 0 && wx < GRIDS)
    v = z[((long long)(b << 10) + hy * GRIDS + wx) * OUTC + c];
  out[idx] = v;
}

// ------------------------- host-side orchestration --------------------------

static inline void launch_gemm(hipStream_t s,
    const float* A, int lda, long long sAw, long long sAh,
    const float* B, int ldb, long long sBw, long long sBh, int bT,
    float* C, int ldc, long long sCw, long long sCh,
    const float* bias, const float* resid, int ldr,
    int M, int N, int K, int batch, int heads, float alpha, int act)
{
  dim3 grid((N + BN - 1) / BN, (M + BM - 1) / BM, batch);
  gemm_bf16_wmma<<<grid, 256, 0, s>>>(A, lda, sAw, sAh, B, ldb, sBw, sBh, bT,
                                      C, ldc, sCw, sCh, bias, resid, ldr,
                                      M, N, K, heads, alpha, act);
}

static inline int ceil_div_ll(long long a, int b) { return (int)((a + b - 1) / b); }

extern "C" void kernel_launch(void* const* d_in, const int* in_sizes, int n_in,
                              void* d_out, int out_size, void* d_ws, size_t ws_size,
                              hipStream_t stream) {
  (void)in_sizes; (void)n_in; (void)out_size; (void)ws_size;

  const float* xin     = (const float*)d_in[0];
  const float* patch_w = (const float*)d_in[1];   // (768, 3,16,16) = N x K
  const float* patch_b = (const float*)d_in[2];
  const float* pos     = (const float*)d_in[3];
  #define BLK(b, j) ((const float*)d_in[4 + (b) * 14 + (j)])
  const float* nk_conv1 = (const float*)d_in[60]; // (256, 768)
  const float* nk_ln1w  = (const float*)d_in[61];
  const float* nk_ln1b  = (const float*)d_in[62];
  const float* nk_conv2 = (const float*)d_in[63]; // (256, 256,3,3)
  const float* nk_ln2w  = (const float*)d_in[64];
  const float* nk_ln2b  = (const float*)d_in[65];
  float* outp = (float*)d_out;

  // workspace layout (f32 elements)
  float* ws = (float*)d_ws;
  float* xb       = ws;                                  // (2048, 768)
  float* yln      = xb   + (long long)NTOK * DD;         // (2048, 768) ; also LN2 out
  float* ywin     = yln  + (long long)NTOK * DD;         // (3528, 768) ; also neck z1
  float* qkv      = ywin + (long long)TWIN * DD;         // (3528, 2304); also im2col bufs
  float* scores   = qkv  + (long long)TWIN * 3 * DD;     // 24*1024*1024 ; also mlp_mid
  float* relh     = scores + (long long)24 * 1024 * 1024;
  float* relw     = relh + (long long)24 * 1024 * 32;
  float* attn_out = relw + (long long)24 * 1024 * 32;    // (3528, 768) ; also neck z1ln
  float* proj_out = attn_out + (long long)TWIN * DD;     // (3528, 768) ; also neck z2

  const long long NE = (long long)NTOK * DD;

  // ---- patch embed: im2col -> GEMM(+bias) -> +pos_embed
  im2col_patch_kernel<<<ceil_div_ll(NE, 256), 256, 0, stream>>>(xin, qkv);
  launch_gemm(stream, qkv, DD, 0, 0, patch_w, DD, 0, 0, /*bT=*/1,
              xb, DD, 0, 0, patch_b, nullptr, 0, NTOK, DD, DD, 1, 1, 1.f, 0);
  add_pos_kernel<<<ceil_div_ll(NE, 256), 256, 0, stream>>>(xb, pos);

  // ---- transformer blocks
  for (int b = 0; b < 4; ++b) {
    const int isGlobal = (b == 3);
    const int S  = isGlobal ? GRIDS : WIN;        // square attention size
    const int L  = S * S;                          // 1024 or 196
    const int T  = isGlobal ? NTOK : TWIN;         // tokens entering attention
    const int NW = isGlobal ? 2 : 18;              // "windows"
    const int Bh = NW * HEADS;
    const long long winStride = (long long)L * 3 * DD;   // qkv per-window stride

    // LN1
    ln_kernel<<<NTOK, 256, 0, stream>>>(xb, yln, BLK(b, 0), BLK(b, 1), DD, 1e-5f, 0);

    const float* tok = yln;
    if (!isGlobal) {
      window_partition_kernel<<<ceil_div_ll((long long)TWIN * DD, 256), 256, 0, stream>>>(yln, ywin);
      tok = ywin;
    }

    // qkv GEMM: (T,768) @ (768,2304) + bias
    launch_gemm(stream, tok, DD, 0, 0, BLK(b, 2), 3 * DD, 0, 0, 0,
                qkv, 3 * DD, 0, 0, BLK(b, 3), nullptr, 0, T, 3 * DD, DD, 1, 1, 1.f, 0);

    // rel-pos terms (use unscaled q)
    {
      long long total = (long long)Bh * L * S;
      rel_kernel<<<ceil_div_ll(total, 256), 256, 0, stream>>>(
          qkv, 3 * DD, winStride, HDIM, HEADS, BLK(b, 6), relh, L, S, S, 1, total);
      rel_kernel<<<ceil_div_ll(total, 256), 256, 0, stream>>>(
          qkv, 3 * DD, winStride, HDIM, HEADS, BLK(b, 7), relw, L, S, S, 0, total);
    }

    // scores = SCALE * q @ k^T  (batched over windows*heads)
    launch_gemm(stream,
                qkv,        3 * DD, winStride, HDIM,            // q
                qkv + DD,   3 * DD, winStride, HDIM, /*bT=*/1,  // k (N x K)
                scores, L, (long long)HEADS * L * L, (long long)L * L,
                nullptr, nullptr, 0, L, L, HDIM, Bh, HEADS, SCALE_ATT, 0);

    // + rel_h + rel_w, softmax (in place)
    softmax_rel_kernel<<<Bh * L, 256, 0, stream>>>(scores, relh, relw, L, S, S);

    // out = attn @ v -> (T, 768) with heads interleaved back
    launch_gemm(stream,
                scores, L, (long long)HEADS * L * L, (long long)L * L,
                qkv + 2 * DD, 3 * DD, winStride, HDIM, 0,
                attn_out, DD, (long long)L * DD, HDIM,
                nullptr, nullptr, 0, L, HDIM, L, Bh, HEADS, 1.f, 0);

    // proj GEMM
    launch_gemm(stream, attn_out, DD, 0, 0, BLK(b, 4), DD, 0, 0, 0,
                proj_out, DD, 0, 0, BLK(b, 5), nullptr, 0, T, DD, DD, 1, 1, 1.f, 0);

    // residual (+ unpartition for windowed blocks)
    unpartition_add_kernel<<<ceil_div_ll(NE, 256), 256, 0, stream>>>(xb, proj_out, !isGlobal);

    // LN2 -> MLP (GELU) -> residual
    ln_kernel<<<NTOK, 256, 0, stream>>>(xb, yln, BLK(b, 8), BLK(b, 9), DD, 1e-5f, 0);
    float* mlp_mid = scores;  // alias (scores no longer needed this block)
    launch_gemm(stream, yln, DD, 0, 0, BLK(b, 10), MLPD, 0, 0, 0,
                mlp_mid, MLPD, 0, 0, BLK(b, 11), nullptr, 0, NTOK, MLPD, DD, 1, 1, 1.f, 1);
    launch_gemm(stream, mlp_mid, MLPD, 0, 0, BLK(b, 12), DD, 0, 0, 0,
                xb, DD, 0, 0, BLK(b, 13), xb, DD, NTOK, DD, MLPD, 1, 1, 1.f, 0);
  }

  // ---- neck (kept token-major (t, C); channel-LN == row LN here)
  float* z1   = ywin;      // (2048, 256)
  float* z1ln = attn_out;  // (2048, 256)
  float* col2 = qkv;       // (2048, 2304)
  float* z2   = proj_out;  // (2048, 256)

  launch_gemm(stream, xb, DD, 0, 0, nk_conv1, DD, 0, 0, /*bT=*/1,
              z1, OUTC, 0, 0, nullptr, nullptr, 0, NTOK, OUTC, DD, 1, 1, 1.f, 0);
  ln_kernel<<<NTOK, 256, 0, stream>>>(z1, z1ln, nk_ln1w, nk_ln1b, OUTC, 1e-6f, 0);
  im2col_neck_kernel<<<ceil_div_ll((long long)NTOK * OUTC * 9, 256), 256, 0, stream>>>(z1ln, col2);
  launch_gemm(stream, col2, OUTC * 9, 0, 0, nk_conv2, OUTC * 9, 0, 0, /*bT=*/1,
              z2, OUTC, 0, 0, nullptr, nullptr, 0, NTOK, OUTC, OUTC * 9, 1, 1, 1.f, 0);
  // final channel-LN, written directly as (B, 256, 32, 32)
  ln_kernel<<<NTOK, 256, 0, stream>>>(z2, outp, nk_ln2w, nk_ln2b, OUTC, 1e-6f, 1);
}
